// GroupedQueryAttention_29231547416705
// MI455X (gfx1250) — compile-verified, matrix operand order checked by reading, not run on hardware
//
#include <hip/hip_runtime.h>
#include <hip/hip_bf16.h>

typedef _Float16 half_t;
typedef __attribute__((ext_vector_type(16))) _Float16 v16h;
typedef __attribute__((ext_vector_type(4)))  _Float16 v4h;
typedef __attribute__((ext_vector_type(8)))  float    v8f;
typedef __attribute__((ext_vector_type(4)))  unsigned int u32x4;
typedef __attribute__((ext_vector_type(8)))  int      i32x8;
typedef __attribute__((ext_vector_type(4)))  int      i32x4;

#define WMMA_F32_16x16x32_F16(A, B, C) \
    __builtin_amdgcn_wmma_f32_16x16x32_f16(false, (A), false, (B), (short)0, (C), false, false)

#if defined(__has_builtin)
#  if __has_builtin(__builtin_amdgcn_tensor_load_to_lds)
#    define HAS_TDM 1
#  else
#    define HAS_TDM 0
#  endif
#else
#  define HAS_TDM 0
#endif

// ---------------------------------------------------------------------------
// Problem constants
// ---------------------------------------------------------------------------
constexpr int HIDDEN   = 2048;
constexpr int NUM_H    = 32;
constexpr int NUM_KV   = 8;
constexpr int HEAD_D   = 64;
constexpr int GROUP    = NUM_H / NUM_KV;   // 4
constexpr int Bsz      = 2;
constexpr int Lseq     = 2048;
constexpr int Mrows    = Bsz * Lseq;       // 4096
constexpr float SCALE  = 0.125f;           // 64^-0.5

// ---------------------------------------------------------------------------
// Tensor Data Mover: 2D f16 tile, global -> LDS, with LDS row padding.
// D# per CDNA5 ISA:
//  group0: [1:0]=count=1 | lds_addr | global_addr | type=2
//  group1: data_size=1(2B), pad_enable, pad_interval/amount codes,
//          tensor_dim0=row_stride, tensor_dim1=tensor_rows,
//          tile_dim0=tile_cols, tile_dim1=tile_rows, dim0_stride=row_stride
//  pad codes: interval c -> 2^(c+1) DWORDs between pads; amount c -> (c+1) DWORDs.
// ---------------------------------------------------------------------------
#if HAS_TDM
__device__ inline void tdm_load_tile_f16(const half_t* __restrict__ g,
                                         unsigned lds_byte_off,
                                         unsigned tile_rows, unsigned tile_cols,
                                         unsigned tensor_rows, unsigned row_stride,
                                         unsigned pad_int_code, unsigned pad_amt_code) {
    const unsigned long long ga = (unsigned long long)(uintptr_t)g;
    u32x4 g0;
    g0[0] = 1u;                                   // count=1 (valid user descriptor)
    g0[1] = lds_byte_off;                         // lds_addr
    g0[2] = (unsigned)ga;                         // global_addr[31:0]
    g0[3] = (unsigned)(ga >> 32) | (2u << 30);    // global_addr[56:32] | type=2
    i32x8 g1;
    g1[0] = (int)((1u << 16)                      // data_size = 2 bytes
                 | (1u << 20)                     // pad_enable
                 | (pad_int_code << 22)
                 | (pad_amt_code << 25));
    g1[1] = (int)((row_stride & 0xFFFFu) << 16);                                  // tensor_dim0 lo
    g1[2] = (int)(((row_stride >> 16) & 0xFFFFu) | ((tensor_rows & 0xFFFFu) << 16)); // dim0 hi | dim1 lo
    g1[3] = (int)(((tensor_rows >> 16) & 0xFFFFu) | ((tile_cols & 0xFFFFu) << 16));  // dim1 hi | tile_dim0
    g1[4] = (int)(tile_rows & 0xFFFFu);                                           // tile_dim1 (tile_dim2=0)
    g1[5] = (int)row_stride;                                                      // dim0_stride[31:0]
    g1[6] = 0;
    g1[7] = 0;
    i32x4 z4 = {0, 0, 0, 0};
#if defined(__clang_major__) && (__clang_major__ >= 23)
    i32x8 z8 = {0, 0, 0, 0, 0, 0, 0, 0};
    __builtin_amdgcn_tensor_load_to_lds(g0, g1, z4, z4, z8, 0);
#else
    __builtin_amdgcn_tensor_load_to_lds(g0, g1, z4, z4, 0);
#endif
}
#endif

__device__ inline unsigned lds_off_of(const void* p) {
    return (unsigned)(uintptr_t)p;   // low 32 bits of an LDS generic address = LDS offset
}

// ---------------------------------------------------------------------------
// WMMA fragment loaders (ISA 7.12.2 layouts, wave32)
// ---------------------------------------------------------------------------
__device__ inline v16h frag_ld_row(const half_t* __restrict__ p, int ld, int lane) {
    const int r     = lane & 15;
    const int khalf = (lane & 16) ? 8 : 0;
    v16h f;
#pragma unroll
    for (int v = 0; v < 8; ++v) {
        const int k = ((v & 4) ? 16 : 0) + khalf + ((v & 3) << 1);
        f[2 * v]     = p[r * ld + k];
        f[2 * v + 1] = p[r * ld + k + 1];
    }
    return f;
}

__device__ inline v16h frag_ld_col(const half_t* __restrict__ p, int ld, int lane) {
    const int n     = lane & 15;
    const int khalf = (lane & 16) ? 8 : 0;
    v16h f;
#pragma unroll
    for (int v = 0; v < 8; ++v) {
        const int k = ((v & 4) ? 16 : 0) + khalf + ((v & 3) << 1);
        f[2 * v]     = p[k * ld + n];
        f[2 * v + 1] = p[(k + 1) * ld + n];
    }
    return f;
}

// ---------------------------------------------------------------------------
// f32 -> f16 bulk convert (4 elems / thread)
// ---------------------------------------------------------------------------
__global__ __launch_bounds__(256)
void gqa_cvt_f16(const float* __restrict__ src, half_t* __restrict__ dst, int n) {
    const int i = (blockIdx.x * 256 + threadIdx.x) * 4;
    if (i < n) {
        const float4 v = *(const float4*)(src + i);
        v4h h;
        h[0] = (half_t)v.x; h[1] = (half_t)v.y; h[2] = (half_t)v.z; h[3] = (half_t)v.w;
        *(v4h*)(dst + i) = h;
    }
}

// ---------------------------------------------------------------------------
// GEMM: C[M,N] = A[M,K] * W[N,K]^T   (f16 in, f16 WMMA, f32 acc)
// 256 thr / 8 waves. Tile 128(M) x 128(N), K-step 32, TDM double-buffered.
// Waves: 4 along M (32 rows) x 2 along N (64 cols) -> 2x4 WMMA frags each.
// Output f32 (Cf) or f16 (Ch) selected by out_f16.
// ---------------------------------------------------------------------------
constexpr int GT_M  = 128;
constexpr int GT_N  = 128;
constexpr int GT_K  = 32;
constexpr int LDS_K = GT_K + 8;   // 40 halves: 64B row + 16B TDM pad

__global__ __launch_bounds__(256)
void gqa_gemm_xwt(const half_t* __restrict__ A, const half_t* __restrict__ W,
                  float* __restrict__ Cf, half_t* __restrict__ Ch,
                  int M, int N, int K, int out_f16) {
    __shared__ half_t sA0[GT_M * LDS_K], sA1[GT_M * LDS_K];
    __shared__ half_t sW0[GT_N * LDS_K], sW1[GT_N * LDS_K];

    const int tid  = threadIdx.x;
    const int lane = tid & 31;
    const int wave = tid >> 5;
    const int bm   = blockIdx.x * GT_M;
    const int bn   = blockIdx.y * GT_N;
    const int wm   = (wave & 3) * 32;    // 4 waves along M
    const int wn   = (wave >> 2) * 64;   // 2 waves along N
    const int nk   = K / GT_K;

#if HAS_TDM
    if (wave == 0) {
        // pad codes: 32 halves = 16 DWORDs -> interval 3; 8 halves = 4 DWORDs -> amount 3
        tdm_load_tile_f16(A + (size_t)bm * K, lds_off_of(sA0), GT_M, GT_K, M, K, 3, 3);
        tdm_load_tile_f16(W + (size_t)bn * K, lds_off_of(sW0), GT_N, GT_K, N, K, 3, 3);
    }
#endif

    v8f acc[2][4];
#pragma unroll
    for (int i = 0; i < 2; ++i)
#pragma unroll
        for (int j = 0; j < 4; ++j) acc[i][j] = (v8f){};

    for (int it = 0; it < nk; ++it) {
        const int cur = it & 1;
        half_t* sA = cur ? sA1 : sA0;
        half_t* sW = cur ? sW1 : sW0;
#if HAS_TDM
        if (wave == 0) {
            if (it + 1 < nk) {
                const size_t knext = (size_t)(it + 1) * GT_K;
                tdm_load_tile_f16(A + (size_t)bm * K + knext,
                                  cur ? lds_off_of(sA0) : lds_off_of(sA1),
                                  GT_M, GT_K, M, K, 3, 3);
                tdm_load_tile_f16(W + (size_t)bn * K + knext,
                                  cur ? lds_off_of(sW0) : lds_off_of(sW1),
                                  GT_N, GT_K, N, K, 3, 3);
                __builtin_amdgcn_s_wait_tensorcnt(2);   // current pair complete
            } else {
                __builtin_amdgcn_s_wait_tensorcnt(0);
            }
        }
#else
        {   // fallback: cooperative copy of the current tiles
            const int k0 = it * GT_K;
#pragma unroll
            for (int i = 0; i < 16; ++i) {
                const int idx = tid + i * 256;           // A: 128x32
                const int r = idx >> 5, c = idx & 31;
                sA[r * LDS_K + c] = A[(size_t)(bm + r) * K + k0 + c];
                sW[r * LDS_K + c] = W[(size_t)(bn + r) * K + k0 + c];
            }
        }
#endif
        __syncthreads();

        const v16h fa0 = frag_ld_row(&sA[(wm +  0) * LDS_K], LDS_K, lane);
        const v16h fa1 = frag_ld_row(&sA[(wm + 16) * LDS_K], LDS_K, lane);
#pragma unroll
        for (int t = 0; t < 4; ++t) {
            const v16h fb = frag_ld_row(&sW[(wn + t * 16) * LDS_K], LDS_K, lane);
            acc[0][t] = WMMA_F32_16x16x32_F16(fa0, fb, acc[0][t]);
            acc[1][t] = WMMA_F32_16x16x32_F16(fa1, fb, acc[1][t]);
        }
        __syncthreads();
    }

    // C fragment: VGPR j -> row j (lanes 0-15) / j+8 (lanes 16-31); col = lane&15
    const int col  = lane & 15;
    const int rofs = (lane & 16) ? 8 : 0;
#pragma unroll
    for (int i = 0; i < 2; ++i) {
#pragma unroll
        for (int j = 0; j < 8; ++j) {
            const size_t r = (size_t)(bm + wm + i * 16 + rofs + j);
#pragma unroll
            for (int t = 0; t < 4; ++t) {
                const size_t cidx = r * N + bn + wn + t * 16 + col;
                if (out_f16) Ch[cidx] = (half_t)acc[i][t][j];
                else         Cf[cidx] = acc[i][t][j];
            }
        }
    }
}

// ---------------------------------------------------------------------------
// RoPE + pack:  src (B, L, H*64) f16  ->  dst (B, H, L, 64) f16
// ---------------------------------------------------------------------------
__global__ __launch_bounds__(256)
void gqa_rope_pack(const half_t* __restrict__ src, half_t* __restrict__ dst,
                   int H, int useRope) {
    const size_t idx = (size_t)blockIdx.x * blockDim.x + threadIdx.x;
    const int d = (int)(idx & 63);
    size_t t = idx >> 6;
    const int l = (int)(t & (Lseq - 1)); t >>= 11;
    const int h = (int)(t % H);
    const int b = (int)(t / H);

    const half_t* row = src + (((size_t)b * Lseq + l) * H + h) * HEAD_D;
    const float x = (float)row[d];
    float out = x;
    if (useRope) {
        const float invf = __powf(10000.0f, -(float)(d >> 1) * (1.0f / 32.0f));
        const float ang  = (float)l * invf;
        float s, c;
        __sincosf(ang, &s, &c);
        const float partner = (d < 32) ? -(float)row[d + 32] : (float)row[d - 32];
        out = c * x + s * partner;
    }
    dst[idx] = (half_t)out;
}

// ---------------------------------------------------------------------------
// Flash attention (causal, GQA). Block = (b, h, 64 q-rows), 4 waves.
// KV streamed in 32-row tiles via TDM, double-buffered.
// ctx written f16 as (B, L, H*64).
// ---------------------------------------------------------------------------
constexpr int LDS_D = HEAD_D + 8;   // 72 halves: 128B row + 16B TDM pad
constexpr int LDS_P = 32 + 8;

__global__ __launch_bounds__(128)
void gqa_attention(const half_t* __restrict__ Qh, const half_t* __restrict__ Kh,
                   const half_t* __restrict__ Vh, half_t* __restrict__ ctx) {
    __shared__ half_t sQ[64 * LDS_D];
    __shared__ half_t sK0[32 * LDS_D], sK1[32 * LDS_D];
    __shared__ half_t sV0[32 * LDS_D], sV1[32 * LDS_D];
    __shared__ half_t sP[4][16 * LDS_P];

    const int tid  = threadIdx.x;
    const int lane = tid & 31;
    const int wave = tid >> 5;
    const int qbase = blockIdx.x * 64;
    const int h = blockIdx.y;
    const int b = blockIdx.z;
    const int hk = h / GROUP;

    const half_t* Qp = Qh + (((size_t)b * NUM_H  + h ) * Lseq + qbase) * HEAD_D;
    const half_t* Kp = Kh + (((size_t)b * NUM_KV + hk) * Lseq) * HEAD_D;
    const half_t* Vp = Vh + (((size_t)b * NUM_KV + hk) * Lseq) * HEAD_D;

    const int nk = (qbase + 64) / 32;   // causal kv-tile count (>= 2)

#if HAS_TDM
    if (wave == 0) {
        // 64-half rows = 32 DWORDs -> interval 4; pad 4 DWORDs -> amount 3
        tdm_load_tile_f16(Qp, lds_off_of(sQ),  64, HEAD_D, Lseq, HEAD_D, 4, 3);
        tdm_load_tile_f16(Kp, lds_off_of(sK0), 32, HEAD_D, Lseq, HEAD_D, 4, 3);
        tdm_load_tile_f16(Vp, lds_off_of(sV0), 32, HEAD_D, Lseq, HEAD_D, 4, 3);
    }
#else
#pragma unroll
    for (int i = 0; i < 32; ++i) {
        const int idx = tid + i * 128;
        const int r = idx >> 6, c = idx & 63;
        sQ[r * LDS_D + c] = Qp[(size_t)r * HEAD_D + c];
    }
#endif

    v8f o0 = {}, o1 = {}, o2 = {}, o3 = {};
    float m_j[8], l_j[8];
#pragma unroll
    for (int j = 0; j < 8; ++j) { m_j[j] = -1e30f; l_j[j] = 0.0f; }

    const int rofs       = (lane & 16) ? 8 : 0;
    const int q_row_base = qbase + wave * 16 + rofs;

    v16h qa0 = {}, qa1 = {};

    for (int it = 0; it < nk; ++it) {
        const int cur = it & 1;
        half_t* sK = cur ? sK1 : sK0;
        half_t* sV = cur ? sV1 : sV0;
#if HAS_TDM
        if (wave == 0) {
            if (it + 1 < nk) {
                const size_t kvn = (size_t)(it + 1) * 32;
                tdm_load_tile_f16(Kp + kvn * HEAD_D, cur ? lds_off_of(sK0) : lds_off_of(sK1),
                                  32, HEAD_D, Lseq, HEAD_D, 4, 3);
                tdm_load_tile_f16(Vp + kvn * HEAD_D, cur ? lds_off_of(sV0) : lds_off_of(sV1),
                                  32, HEAD_D, Lseq, HEAD_D, 4, 3);
                __builtin_amdgcn_s_wait_tensorcnt(2);   // Q + current K/V complete
            } else {
                __builtin_amdgcn_s_wait_tensorcnt(0);
            }
        }
#else
        {
            const int kv0 = it * 32;
#pragma unroll
            for (int i = 0; i < 16; ++i) {
                const int idx = tid + i * 128;
                const int r = idx >> 6, c = idx & 63;
                sK[r * LDS_D + c] = Kp[(size_t)(kv0 + r) * HEAD_D + c];
                sV[r * LDS_D + c] = Vp[(size_t)(kv0 + r) * HEAD_D + c];
            }
        }
#endif
        __syncthreads();

        if (it == 0) {   // Q resident after first wait
            qa0 = frag_ld_row(&sQ[(wave * 16) * LDS_D],      LDS_D, lane);
            qa1 = frag_ld_row(&sQ[(wave * 16) * LDS_D + 32], LDS_D, lane);
        }

        // S = Q (16x64) . K^T (64x32) -> two 16x16 tiles (D split in two k=32 chunks)
        const v16h k0f = frag_ld_row(&sK[0],               LDS_D, lane);
        const v16h k0b = frag_ld_row(&sK[32],              LDS_D, lane);
        const v16h k1f = frag_ld_row(&sK[16 * LDS_D],      LDS_D, lane);
        const v16h k1b = frag_ld_row(&sK[16 * LDS_D + 32], LDS_D, lane);
        v8f s0 = {}, s1 = {};
        s0 = WMMA_F32_16x16x32_F16(qa0, k0f, s0);
        s0 = WMMA_F32_16x16x32_F16(qa1, k0b, s0);
        s1 = WMMA_F32_16x16x32_F16(qa0, k1f, s1);
        s1 = WMMA_F32_16x16x32_F16(qa1, k1b, s1);

        // scale + causal mask + online softmax
        const int kv0  = it * 32;
        const int kvc0 = kv0 + (lane & 15);
        const int kvc1 = kvc0 + 16;
#pragma unroll
        for (int j = 0; j < 8; ++j) {
            const int qr = q_row_base + j;
            s0[j] = (kvc0 <= qr) ? s0[j] * SCALE : -1e30f;
            s1[j] = (kvc1 <= qr) ? s1[j] * SCALE : -1e30f;
        }
#pragma unroll
        for (int j = 0; j < 8; ++j) {
            float mx = fmaxf(s0[j], s1[j]);
#pragma unroll
            for (int msk = 1; msk < 16; msk <<= 1)
                mx = fmaxf(mx, __shfl_xor(mx, msk, 32));
            const float mnew = fmaxf(m_j[j], mx);
            const float corr = __expf(m_j[j] - mnew);
            m_j[j] = mnew;
            const float p0 = __expf(s0[j] - mnew);
            const float p1 = __expf(s1[j] - mnew);
            s0[j] = p0; s1[j] = p1;
            float rs = p0 + p1;
#pragma unroll
            for (int msk = 1; msk < 16; msk <<= 1)
                rs += __shfl_xor(rs, msk, 32);
            l_j[j] = l_j[j] * corr + rs;
            o0[j] *= corr; o1[j] *= corr; o2[j] *= corr; o3[j] *= corr;
        }

        // P (C-layout) -> LDS -> A-fragment layout
        half_t* Pp = &sP[wave][0];
#pragma unroll
        for (int j = 0; j < 8; ++j) {
            const int r = j + rofs;
            Pp[r * LDS_P +      (lane & 15)] = (half_t)s0[j];
            Pp[r * LDS_P + 16 + (lane & 15)] = (half_t)s1[j];
        }
        __syncthreads();

        const v16h pa  = frag_ld_row(Pp, LDS_P, lane);        // P 16x32
        const v16h vb0 = frag_ld_col(&sV[ 0], LDS_D, lane);
        const v16h vb1 = frag_ld_col(&sV[16], LDS_D, lane);
        const v16h vb2 = frag_ld_col(&sV[32], LDS_D, lane);
        const v16h vb3 = frag_ld_col(&sV[48], LDS_D, lane);
        o0 = WMMA_F32_16x16x32_F16(pa, vb0, o0);
        o1 = WMMA_F32_16x16x32_F16(pa, vb1, o1);
        o2 = WMMA_F32_16x16x32_F16(pa, vb2, o2);
        o3 = WMMA_F32_16x16x32_F16(pa, vb3, o3);

        __syncthreads();   // protect buf[cur] before TDM refill in it+1
    }

    // Normalize + write ctx (B, L, H*64) f16
    const int col = lane & 15;
#pragma unroll
    for (int j = 0; j < 8; ++j) {
        const float inv = 1.0f / l_j[j];
        const int qr = q_row_base + j;
        half_t* dst = ctx + ((size_t)b * Lseq + qr) * HIDDEN + h * HEAD_D;
        dst[      col] = (half_t)(o0[j] * inv);
        dst[16 + col] = (half_t)(o1[j] * inv);
        dst[32 + col] = (half_t)(o2[j] * inv);
        dst[48 + col] = (half_t)(o3[j] * inv);
    }
}

// ---------------------------------------------------------------------------
// Launch
// ---------------------------------------------------------------------------
extern "C" void kernel_launch(void* const* d_in, const int* in_sizes, int n_in,
                              void* d_out, int out_size, void* d_ws, size_t ws_size,
                              hipStream_t stream) {
    const float* hs = (const float*)d_in[0];
    // d_in[1] = attention_mask (pure causal, applied analytically)
    // d_in[2] = position_ids   (== arange(L), applied analytically)
    const float* wq = (const float*)d_in[3];
    const float* wk = (const float*)d_in[4];
    const float* wv = (const float*)d_in[5];
    const float* wo = (const float*)d_in[6];
    float* out = (float*)d_out;

    // Workspace (bytes). ctx_h aliases q_lin (q_lin fully consumed by rope_pack
    // before attention writes ctx; stream-ordered so safe).
    char* ws = (char*)d_ws;
    half_t* hs_h  = (half_t*)(ws);                        // (B,L,2048)     16.8 MB
    half_t* wq_h  = (half_t*)(ws + 16777216);             // (2048,2048)     8.4 MB
    half_t* wk_h  = (half_t*)(ws + 25165824);             // ( 512,2048)     2.1 MB
    half_t* wv_h  = (half_t*)(ws + 27262976);             // ( 512,2048)     2.1 MB
    half_t* wo_h  = (half_t*)(ws + 29360128);             // (2048,2048)     8.4 MB
    half_t* q_lin = (half_t*)(ws + 37748736);             // (B,L,2048)     16.8 MB
    half_t* k_lin = (half_t*)(ws + 54525952);             // (B,L, 512)      4.2 MB
    half_t* v_lin = (half_t*)(ws + 58720256);             // (B,L, 512)      4.2 MB
    half_t* qh    = (half_t*)(ws + 62914560);             // (B,32,L,64)    16.8 MB
    half_t* kh    = (half_t*)(ws + 79691776);             // (B, 8,L,64)     4.2 MB
    half_t* vh    = (half_t*)(ws + 83886080);             // (B, 8,L,64)     4.2 MB
    half_t* ctx_h = q_lin;                                // alias

    // 0) One-time f32 -> f16 conversion of all GEMM operands
    const int nHS = Mrows * HIDDEN;            // 8,388,608
    const int nWQ = HIDDEN * HIDDEN;           // 4,194,304
    const int nWK = NUM_KV * HEAD_D * HIDDEN;  // 1,048,576
    gqa_cvt_f16<<<nHS / 1024, 256, 0, stream>>>(hs, hs_h, nHS);
    gqa_cvt_f16<<<nWQ / 1024, 256, 0, stream>>>(wq, wq_h, nWQ);
    gqa_cvt_f16<<<nWK / 1024, 256, 0, stream>>>(wk, wk_h, nWK);
    gqa_cvt_f16<<<nWK / 1024, 256, 0, stream>>>(wv, wv_h, nWK);
    gqa_cvt_f16<<<nWQ / 1024, 256, 0, stream>>>(wo, wo_h, nWQ);

    // 1) QKV projections (f16 out)
    dim3 gq(Mrows / GT_M, HIDDEN / GT_N);               // 32 x 16
    dim3 gkv(Mrows / GT_M, (NUM_KV * HEAD_D) / GT_N);   // 32 x 4
    gqa_gemm_xwt<<<gq,  256, 0, stream>>>(hs_h, wq_h, nullptr, q_lin, Mrows, HIDDEN, HIDDEN, 1);
    gqa_gemm_xwt<<<gkv, 256, 0, stream>>>(hs_h, wk_h, nullptr, k_lin, Mrows, NUM_KV * HEAD_D, HIDDEN, 1);
    gqa_gemm_xwt<<<gkv, 256, 0, stream>>>(hs_h, wv_h, nullptr, v_lin, Mrows, NUM_KV * HEAD_D, HIDDEN, 1);

    // 2) RoPE + pack to head-major f16
    const size_t nQ = (size_t)Bsz * NUM_H  * Lseq * HEAD_D;   // 8,388,608
    const size_t nK = (size_t)Bsz * NUM_KV * Lseq * HEAD_D;   // 2,097,152
    gqa_rope_pack<<<(unsigned)(nQ / 256), 256, 0, stream>>>(q_lin, qh, NUM_H, 1);
    gqa_rope_pack<<<(unsigned)(nK / 256), 256, 0, stream>>>(k_lin, kh, NUM_KV, 1);
    gqa_rope_pack<<<(unsigned)(nK / 256), 256, 0, stream>>>(v_lin, vh, NUM_KV, 0);

    // 3) Causal flash attention (ctx f16)
    dim3 ga(Lseq / 64, NUM_H, Bsz);                     // 32 x 32 x 2
    gqa_attention<<<ga, 128, 0, stream>>>(qh, kh, vh, ctx_h);

    // 4) Output projection (f32 out)
    gqa_gemm_xwt<<<gq, 256, 0, stream>>>(ctx_h, wo_h, out, nullptr, Mrows, HIDDEN, HIDDEN, 0);
}
